// Attention_70162585747722
// MI455X (gfx1250) — matrix operand order repaired and checked by reading, not compiled, not run on hardware
//
#include <hip/hip_runtime.h>
#include <hip/hip_bf16.h>
#include <cstdint>

// ---------------------------------------------------------------------------
// Types for CDNA5 WMMA (wave32)
// ---------------------------------------------------------------------------
typedef __attribute__((ext_vector_type(16))) __bf16 v16bf;
typedef __attribute__((ext_vector_type(8)))  float  v8f;

union FragBF16 {
    uint4  u[2];
    v16bf  v;
};

// Problem constants
#define BB 4
#define NN 2048
#define DD 1024
#define BN (BB * NN)   // 8192

// GEMM tiling
#define TM 128
#define TN 128
#define KC 64
#define LDSPAD 8       // 16-byte row pad to break LDS bank conflicts

// flags
#define F_CAUSAL 1
#define F_OUTF32 2
#define F_TRANS  4
#define F_BIAS   8

// ---------------------------------------------------------------------------
// CDNA5 async global->LDS copy (ASYNCcnt) with fallback
// ---------------------------------------------------------------------------
#if defined(__HIP_DEVICE_COMPILE__) && defined(__gfx1250__) && \
    __has_builtin(__builtin_amdgcn_global_load_async_to_lds_b128)
#define USE_ASYNC 1
#else
#define USE_ASYNC 0
#endif

// Builtin parameter type per hipcc diagnostic:
//   '__attribute__((__vector_size__(4 * sizeof(int)))) int __device__ *'
// i.e. a vector_size(16) int pointer in AS1 (global) / AS3 (LDS).
typedef int v4i_vs __attribute__((vector_size(4 * sizeof(int))));
#define TO_GLOBAL(p) ((__attribute__((address_space(1))) v4i_vs*)(uintptr_t)(p))
#define TO_LDS(p)    ((__attribute__((address_space(3))) v4i_vs*)(uint32_t)(uintptr_t)(p))

// ---------------------------------------------------------------------------
// fp32 -> bf16 convert (grid-stride)
// ---------------------------------------------------------------------------
__global__ __launch_bounds__(256)
void convert_f32_bf16(const float* __restrict__ x, __hip_bfloat16* __restrict__ y,
                      size_t n) {
    size_t i = (size_t)blockIdx.x * blockDim.x + threadIdx.x;
    size_t stride = (size_t)gridDim.x * blockDim.x;
    for (; i < n; i += stride) y[i] = __float2bfloat16(x[i]);
}

// ---------------------------------------------------------------------------
// Weight transpose + convert: T[o*D + i] = bf16(W[i*D + o]).  blockIdx.z picks W.
// ---------------------------------------------------------------------------
__global__ __launch_bounds__(256)
void transpose_weights(const float* __restrict__ W0, const float* __restrict__ W1,
                       const float* __restrict__ W2, const float* __restrict__ W3,
                       __hip_bfloat16* __restrict__ T0, __hip_bfloat16* __restrict__ T1,
                       __hip_bfloat16* __restrict__ T2, __hip_bfloat16* __restrict__ T3,
                       int D) {
    __shared__ float tile[32][33];
    const float* W; __hip_bfloat16* T;
    switch (blockIdx.z) {
        case 0: W = W0; T = T0; break;
        case 1: W = W1; T = T1; break;
        case 2: W = W2; T = T2; break;
        default: W = W3; T = T3; break;
    }
    const int bx = blockIdx.x * 32;   // o-block
    const int by = blockIdx.y * 32;   // i-block
    const int tx = threadIdx.x;       // 0..31
    const int ty = threadIdx.y;       // 0..7
#pragma unroll
    for (int j = 0; j < 32; j += 8)
        tile[ty + j][tx] = W[(size_t)(by + ty + j) * D + (bx + tx)];
    __syncthreads();
#pragma unroll
    for (int j = 0; j < 32; j += 8)
        T[(size_t)(bx + ty + j) * D + (by + tx)] = __float2bfloat16(tile[tx][ty + j]);
}

// ---------------------------------------------------------------------------
// Generic bf16 GEMM with f32 accumulation via v_wmma_f32_16x16x32_bf16.
//   C[M x N] = A[M x K] * B, where B is supplied transposed: Bt[N x K].
// 256 threads = 8 waves (2x4).  Per wave: 64x32 output = 4x2 WMMA tiles.
// Double-buffered LDS fed by GLOBAL_LOAD_ASYNC_TO_LDS_B128 (ASYNCcnt),
// prefetching tile t+1 while tile t is consumed by WMMA.
// strideA/strideB in bf16 elements, strideC in BYTES (C may be f32 or bf16).
// ---------------------------------------------------------------------------
__global__ __launch_bounds__(256)
void gemm_bf16_wmma(const __hip_bfloat16* __restrict__ A, long long strideA, int lda,
                    const __hip_bfloat16* __restrict__ Bt, long long strideB, int ldb,
                    void* __restrict__ C, long long strideC, int ldc,
                    const float* __restrict__ bias, float scale,
                    int M, int N, int K, int flags) {
#if USE_ASYNC
    __shared__ __align__(16) __hip_bfloat16 As[2][TM][KC + LDSPAD];
    __shared__ __align__(16) __hip_bfloat16 Bs[2][TN][KC + LDSPAD];
#else
    __shared__ __align__(16) __hip_bfloat16 As[1][TM][KC + LDSPAD];
    __shared__ __align__(16) __hip_bfloat16 Bs[1][TN][KC + LDSPAD];
#endif

    const int z = blockIdx.z;
    A  += (size_t)z * strideA;
    Bt += (size_t)z * strideB;
    char* Cb = (char*)C + (size_t)z * strideC;

    const int rowBase = blockIdx.y * TM;   // per-batch row index
    const int colBase = blockIdx.x * TN;   // per-batch col index

    const int tid  = threadIdx.x;
    const int lane = tid & 31;
    const int wid  = tid >> 5;
    const int wm   = wid >> 2;             // 0..1
    const int wn   = wid & 3;              // 0..3
    const int half = lane >> 4;            // 0 or 1
    const int lr   = lane & 15;            // 0..15

    v8f acc[4][2];
#pragma unroll
    for (int mt = 0; mt < 4; ++mt)
#pragma unroll
        for (int nt = 0; nt < 2; ++nt)
            acc[mt][nt] = (v8f){0.f, 0.f, 0.f, 0.f, 0.f, 0.f, 0.f, 0.f};

    // Consume one staged LDS buffer: 12 ds_load_b128 fragment loads, 8 WMMAs/step.
    auto compute = [&](int buf) {
#pragma unroll
        for (int ks = 0; ks < 2; ++ks) {
            const int k0 = ks * 32;
            FragBF16 a[4], b[2];
#pragma unroll
            for (int mt = 0; mt < 4; ++mt) {
                const int r  = wm * 64 + mt * 16 + lr;
                const int kk = k0 + half * 8;        // A layout: K0..7 | K16..23
                a[mt].u[0] = *(const uint4*)&As[buf][r][kk];
                a[mt].u[1] = *(const uint4*)&As[buf][r][kk + 16];
            }
#pragma unroll
            for (int nt = 0; nt < 2; ++nt) {
                const int cc = wn * 32 + nt * 16 + lr;
                const int kk = k0 + half * 8;        // B layout: K0..7 | K16..23
                b[nt].u[0] = *(const uint4*)&Bs[buf][cc][kk];
                b[nt].u[1] = *(const uint4*)&Bs[buf][cc][kk + 16];
            }
#pragma unroll
            for (int mt = 0; mt < 4; ++mt)
#pragma unroll
                for (int nt = 0; nt < 2; ++nt)
                    acc[mt][nt] = __builtin_amdgcn_wmma_f32_16x16x32_bf16(
                        false, a[mt].v, false, b[nt].v,
                        (short)0, acc[mt][nt], false, false);
        }
    };

    // Causal fast path: tile entirely above diagonal -> skip all math.
    const bool fullmask = (flags & F_CAUSAL) && (colBase > rowBase);

    if (!fullmask) {
        const int ldCol = (tid & 7) * 8;   // 0..56, 16B chunks along K
        const int ldRow = tid >> 3;        // 0..31
        const int nk = K / KC;
#if USE_ASYNC
        // Async-copy one KC-deep tile pair into LDS buffer `buf` (8 x b128/thread).
        auto issue_copies = [&](int buf, int kb) {
#pragma unroll
            for (int i = 0; i < 4; ++i) {
                const int r = ldRow + i * 32;
                __builtin_amdgcn_global_load_async_to_lds_b128(
                    TO_GLOBAL(A + (size_t)(rowBase + r) * lda + kb + ldCol),
                    TO_LDS(&As[buf][r][ldCol]), 0, 0);
                __builtin_amdgcn_global_load_async_to_lds_b128(
                    TO_GLOBAL(Bt + (size_t)(colBase + r) * ldb + kb + ldCol),
                    TO_LDS(&Bs[buf][r][ldCol]), 0, 0);
            }
        };
        issue_copies(0, 0);
        for (int t = 0; t < nk; ++t) {
            const int cur = t & 1;
            if (t + 1 < nk) {
                issue_copies(cur ^ 1, (t + 1) * KC);
                // 16 outstanding, in-order completion: <=8 left means tile t done.
                asm volatile("s_wait_asynccnt 0x8" ::: "memory");
            } else {
                asm volatile("s_wait_asynccnt 0x0" ::: "memory");
            }
            __syncthreads();      // tile t visible to all waves
            compute(cur);
            __syncthreads();      // all waves done reading buf before overwrite
        }
#else
        // Fallback: batch global loads into registers (all in flight), then LDS.
        for (int t = 0; t < nk; ++t) {
            const int kb = t * KC;
            uint4 ra[4], rb[4];
#pragma unroll
            for (int i = 0; i < 4; ++i) {
                const int r = ldRow + i * 32;
                ra[i] = *(const uint4*)(A + (size_t)(rowBase + r) * lda + kb + ldCol);
                rb[i] = *(const uint4*)(Bt + (size_t)(colBase + r) * ldb + kb + ldCol);
            }
#pragma unroll
            for (int i = 0; i < 4; ++i) {
                const int r = ldRow + i * 32;
                *(uint4*)&As[0][r][ldCol] = ra[i];
                *(uint4*)&Bs[0][r][ldCol] = rb[i];
            }
            __syncthreads();
            compute(0);
            __syncthreads();
        }
#endif
    }

    // Epilogue: bias, scale, causal mask, store (f32 / bf16 / bf16-transposed).
#pragma unroll
    for (int mt = 0; mt < 4; ++mt) {
#pragma unroll
        for (int nt = 0; nt < 2; ++nt) {
            const int col = colBase + wn * 32 + nt * 16 + lr;
#pragma unroll
            for (int r = 0; r < 8; ++r) {
                const int row = rowBase + wm * 64 + mt * 16 + r + half * 8;
                float val = acc[mt][nt][r];
                if (flags & F_BIAS) val += bias[col];
                val *= scale;
                if ((flags & F_CAUSAL) && (col > row)) val = -3.0e38f;
                if (flags & F_OUTF32) {
                    ((float*)Cb)[(size_t)row * ldc + col] = val;
                } else if (flags & F_TRANS) {
                    ((__hip_bfloat16*)Cb)[(size_t)col * ldc + row] = __float2bfloat16(val);
                } else {
                    ((__hip_bfloat16*)Cb)[(size_t)row * ldc + col] = __float2bfloat16(val);
                }
            }
        }
    }
}

// ---------------------------------------------------------------------------
// Row softmax over S (f32, rowlen=2048), writes P (bf16) aliased over S buffer.
// One 256-thread block per row; each thread owns 8 elements. All S reads of the
// row happen before any P write (same block, syncthreads between).
// ---------------------------------------------------------------------------
__global__ __launch_bounds__(256)
void softmax_rows(float* __restrict__ S, int rowlen) {
    const size_t row = blockIdx.x;
    float* Sr = S + row * (size_t)rowlen;
    __hip_bfloat16* Pr = (__hip_bfloat16*)S + row * (size_t)(2 * rowlen);
    const int t = threadIdx.x;

    float v[8];
    float m = -3.4e38f;
#pragma unroll
    for (int j = 0; j < 8; ++j) {
        v[j] = Sr[t + j * 256];
        m = fmaxf(m, v[j]);
    }
    __shared__ float red[256];
    red[t] = m;
    __syncthreads();
    for (int s = 128; s > 0; s >>= 1) {
        if (t < s) red[t] = fmaxf(red[t], red[t + s]);
        __syncthreads();
    }
    m = red[0];
    __syncthreads();

    float l = 0.f;
#pragma unroll
    for (int j = 0; j < 8; ++j) {
        v[j] = __expf(v[j] - m);
        l += v[j];
    }
    red[t] = l;
    __syncthreads();
    for (int s = 128; s > 0; s >>= 1) {
        if (t < s) red[t] += red[t + s];
        __syncthreads();
    }
    const float inv = 1.0f / red[0];
#pragma unroll
    for (int j = 0; j < 8; ++j)
        Pr[t + j * 256] = __float2bfloat16(v[j] * inv);
}

// ---------------------------------------------------------------------------
// Host-side orchestration
// ---------------------------------------------------------------------------
extern "C" void kernel_launch(void* const* d_in, const int* in_sizes, int n_in,
                              void* d_out, int out_size, void* d_ws, size_t ws_size,
                              hipStream_t stream) {
    (void)in_sizes; (void)n_in; (void)out_size; (void)ws_size;

    const float* x  = (const float*)d_in[0];
    // d_in[1] = mask (causal triu, recomputed in-kernel; unused)
    const float* Wq = (const float*)d_in[2];
    const float* bq = (const float*)d_in[3];
    const float* Wk = (const float*)d_in[4];
    const float* bk = (const float*)d_in[5];
    const float* Wv = (const float*)d_in[6];
    const float* bv = (const float*)d_in[7];
    const float* Wo = (const float*)d_in[8];
    const float* bo = (const float*)d_in[9];
    float* out = (float*)d_out;

    // workspace carve (256B aligned)
    char* p = (char*)d_ws;
    auto carve = [&](size_t bytes) -> void* {
        void* r = (void*)p;
        p += (bytes + 255) & ~(size_t)255;
        return r;
    };
    __hip_bfloat16* xb  = (__hip_bfloat16*)carve((size_t)BN * DD * 2);
    __hip_bfloat16* Wqt = (__hip_bfloat16*)carve((size_t)DD * DD * 2);
    __hip_bfloat16* Wkt = (__hip_bfloat16*)carve((size_t)DD * DD * 2);
    __hip_bfloat16* Wvt = (__hip_bfloat16*)carve((size_t)DD * DD * 2);
    __hip_bfloat16* Wot = (__hip_bfloat16*)carve((size_t)DD * DD * 2);
    __hip_bfloat16* Qb  = (__hip_bfloat16*)carve((size_t)BN * DD * 2);
    __hip_bfloat16* Kb  = (__hip_bfloat16*)carve((size_t)BN * DD * 2);
    __hip_bfloat16* Vt  = (__hip_bfloat16*)carve((size_t)BN * DD * 2); // [b][d][n]
    __hip_bfloat16* Ab  = (__hip_bfloat16*)carve((size_t)BN * DD * 2);
    float*          S   = (float*)carve((size_t)BB * NN * NN * 4);     // P bf16 aliased

    const float qscale = 0.03125f; // 1/sqrt(1024)

    // 1) x -> bf16
    convert_f32_bf16<<<4096, 256, 0, stream>>>(x, xb, (size_t)BN * DD);

    // 2) transpose + convert all 4 weight matrices
    transpose_weights<<<dim3(DD / 32, DD / 32, 4), dim3(32, 8), 0, stream>>>(
        Wq, Wk, Wv, Wo, Wqt, Wkt, Wvt, Wot, DD);

    // 3) Q = (x*Wq + bq) * 1/sqrt(D)   [8192 x 1024], flattened batch
    gemm_bf16_wmma<<<dim3(DD / TN, BN / TM, 1), 256, 0, stream>>>(
        xb, 0, DD, Wqt, 0, DD, Qb, 0, DD, bq, qscale, BN, DD, DD, F_BIAS);

    // 4) K = x*Wk + bk                 [8192 x 1024]
    gemm_bf16_wmma<<<dim3(DD / TN, BN / TM, 1), 256, 0, stream>>>(
        xb, 0, DD, Wkt, 0, DD, Kb, 0, DD, bk, 1.0f, BN, DD, DD, F_BIAS);

    // 5) V = x*Wv + bv, stored TRANSPOSED per batch: Vt[b][d][n]
    gemm_bf16_wmma<<<dim3(DD / TN, NN / TM, BB), 256, 0, stream>>>(
        xb, (long long)NN * DD, DD, Wvt, 0, DD,
        Vt, (long long)DD * NN * 2, NN, bv, 1.0f, NN, DD, DD, F_BIAS | F_TRANS);

    // 6) S = Q * K^T with causal mask, f32 out  [b: 2048 x 2048], K-dim = 1024
    gemm_bf16_wmma<<<dim3(NN / TN, NN / TM, BB), 256, 0, stream>>>(
        Qb, (long long)NN * DD, DD, Kb, (long long)NN * DD, DD,
        S, (long long)NN * NN * 4, NN, nullptr, 1.0f, NN, NN, DD,
        F_CAUSAL | F_OUTF32);

    // 7) row softmax -> P (bf16, aliased over S, row pitch 4096 bf16)
    softmax_rows<<<BB * NN, 256, 0, stream>>>(S, NN);

    // 8) attn = P * V  (A = P [2048 x 2048] pitch 4096, Bt = Vt [1024 x 2048])
    gemm_bf16_wmma<<<dim3(DD / TN, NN / TM, BB), 256, 0, stream>>>(
        (const __hip_bfloat16*)S, (long long)NN * 2 * NN, 2 * NN,
        Vt, (long long)DD * NN, NN,
        Ab, (long long)NN * DD * 2, DD, nullptr, 1.0f, NN, DD, NN, 0);

    // 9) out = attn * Wo + bo  -> f32   [8192 x 1024], flattened batch
    gemm_bf16_wmma<<<dim3(DD / TN, BN / TM, 1), 256, 0, stream>>>(
        Ab, 0, DD, Wot, 0, DD, out, 0, DD, bo, 1.0f, BN, DD, DD,
        F_BIAS | F_OUTF32);
}
